// SuperExpertMoE_53772990546013
// MI455X (gfx1250) — matrix operand order repaired and checked by reading, not compiled, not run on hardware
//
#include <hip/hip_runtime.h>
#include <hip/hip_bf16.h>

// ---------------- problem constants ----------------
constexpr int BB = 2, TT = 2048, CC = 2048;
constexpr int EE = 8, FF = 1408;
constexpr int NN = BB * TT;                 // 4096 tokens
constexpr float AUX_COEF = 0.01f, Z_COEF = 0.001f;

// ---------------- GEMM tile constants ----------------
constexpr int TM = 128, TN = 128, TK = 64;  // block tile (bf16)
constexpr int PITCH = TK + 8;               // LDS row pitch (ushorts): 144B rows break bank stride

typedef __attribute__((ext_vector_type(16))) __bf16 v16bf;
typedef __attribute__((ext_vector_type(8)))  float  v8f;
typedef __attribute__((__vector_size__(16))) int    vi4;   // 16B vector for async b128 builtin

struct Frag32 { uint4 lo, hi; };            // 32 bytes == v16bf

// ---------------- CDNA5 async global->LDS path (guarded) ----------------
#if defined(__has_builtin)
#if __has_builtin(__builtin_amdgcn_global_load_async_to_lds_b128)
#define HAS_ASYNC_LDS 1
#endif
#endif
#ifndef HAS_ASYNC_LDS
#define HAS_ASYNC_LDS 0
#endif

__device__ __forceinline__ void async_cp16(const unsigned short* src, unsigned short* dst) {
#if HAS_ASYNC_LDS
    // signature (from compiler diagnostic): (int4 AS1* gsrc, int4 AS3* ldst, imm offset, imm cpol)
    __builtin_amdgcn_global_load_async_to_lds_b128(
        (__attribute__((address_space(1))) vi4*)(void*)const_cast<unsigned short*>(src),
        (__attribute__((address_space(3))) vi4*)(void*)dst,
        0, 0);
#else
    *(uint4*)dst = *(const uint4*)src;      // sync fallback
#endif
}
__device__ __forceinline__ void wait_async_le8() {
#if HAS_ASYNC_LDS
#if __has_builtin(__builtin_amdgcn_s_wait_asynccnt)
    __builtin_amdgcn_s_wait_asynccnt(8);
#else
    asm volatile("s_wait_asynccnt 8" ::: "memory");
#endif
#endif
}
__device__ __forceinline__ void wait_async_0() {
#if HAS_ASYNC_LDS
#if __has_builtin(__builtin_amdgcn_s_wait_asynccnt)
    __builtin_amdgcn_s_wait_asynccnt(0);
#else
    asm volatile("s_wait_asynccnt 0" ::: "memory");
#endif
#endif
}

// fp32 -> bf16 round-to-nearest-even (bit trick)
__device__ __forceinline__ unsigned short f2bf(float f) {
    unsigned u = __float_as_uint(f);
    u += 0x7FFFu + ((u >> 16) & 1u);
    return (unsigned short)(u >> 16);
}
__device__ __forceinline__ unsigned pkbf(float lo, float hi) {
    return (unsigned)f2bf(lo) | ((unsigned)f2bf(hi) << 16);
}
__device__ __forceinline__ uint4 cvt8(float4 a, float4 b) {
    return make_uint4(pkbf(a.x, a.y), pkbf(a.z, a.w), pkbf(b.x, b.y), pkbf(b.z, b.w));
}

// A fragment: 16x32 bf16. Lanes 0-15: M=lane, K={0..7,16..23}; lanes 16-31: K={8..15,24..31}.
__device__ __forceinline__ v16bf ld_a(const unsigned short* s, int row, int ks, int laneHalf) {
    const unsigned short* p = s + row * PITCH + ks + laneHalf * 8;
    Frag32 f;
    f.lo = *(const uint4*)(p);
    f.hi = *(const uint4*)(p + 16);
    return __builtin_bit_cast(v16bf, f);
}
// B fragment: 32x16 bf16, K-innermost in LDS: lanes 0-15 K={0..7,16..23}, lanes 16-31 K={8..15,24..31}.
__device__ __forceinline__ v16bf ld_b(const unsigned short* s, int nrow, int ks, int laneHalf) {
    const unsigned short* p = s + nrow * PITCH + ks + laneHalf * 8;
    Frag32 f;
    f.lo = *(const uint4*)(p);
    f.hi = *(const uint4*)(p + 16);
    return __builtin_bit_cast(v16bf, f);
}
__device__ __forceinline__ v8f wmma_bf16(v16bf a, v16bf b, v8f c) {
    return __builtin_amdgcn_wmma_f32_16x16x32_bf16(false, a, false, b, (short)0, c, false, false);
}

// stage one 128x64 bf16 A tile (optionally row-gathered) + one 128x64 B tile into LDS
__device__ __forceinline__ void stage_tile(
    unsigned short* sA, unsigned short* sB, int t,
    const unsigned short* Amat, int lda, const int* gather, int n_e,
    const unsigned short* Bmat, int ldb, const unsigned short* zbuf,
    int tileM, int tileN, int kb)
{
#pragma unroll
    for (int i = 0; i < 4; ++i) {
        int slot = t + i * 256;
        int row = slot >> 3, seg = slot & 7;          // 128 rows x 8 chunks of 16B
        int grow = tileM * TM + row;
        const unsigned short* srcA = zbuf;
        if (grow < n_e) {
            int tok = gather ? gather[grow] : grow;
            srcA = Amat + (size_t)tok * lda + kb + seg * 8;
        }
        async_cp16(srcA, sA + row * PITCH + seg * 8);
        const unsigned short* srcB = Bmat + (size_t)(tileN * TN + row) * ldb + kb + seg * 8;
        async_cp16(srcB, sB + row * PITCH + seg * 8);
    }
}

// 8-wave 128x128 tile compute: each wave 32x64 = 2x4 wmma accumulators, two k-steps per LDS tile
__device__ __forceinline__ void compute_tile(
    const unsigned short* sA, const unsigned short* sB, v8f acc[2][4],
    int waveM, int waveN, int laneM, int laneHalf)
{
#pragma unroll
    for (int ks = 0; ks < TK; ks += 32) {
        v16bf af[2], bf[4];
#pragma unroll
        for (int mi = 0; mi < 2; ++mi) af[mi] = ld_a(sA, waveM * 32 + mi * 16 + laneM, ks, laneHalf);
#pragma unroll
        for (int ni = 0; ni < 4; ++ni) bf[ni] = ld_b(sB, waveN * 64 + ni * 16 + laneM, ks, laneHalf);
#pragma unroll
        for (int mi = 0; mi < 2; ++mi)
#pragma unroll
            for (int ni = 0; ni < 4; ++ni)
                acc[mi][ni] = wmma_bf16(af[mi], bf[ni], acc[mi][ni]);
    }
}

// ================= fp32 -> bf16 conversion pass (streaming, once per launch) =================
__global__ __launch_bounds__(256) void cvt_bf16_kernel(
    const float* __restrict__ src, unsigned short* __restrict__ dst)
{
    size_t i = (size_t)blockIdx.x * 256 + threadIdx.x;   // 8 floats per thread
    const float4* s4 = (const float4*)src;
    ((uint4*)dst)[i] = cvt8(s4[2 * i], s4[2 * i + 1]);
}

// ================= gate: logits, softmax, top-2, routing lists, losses =================
__global__ __launch_bounds__(256) void moe_gate_kernel(
    const float* __restrict__ x, const float* __restrict__ gw,
    float* __restrict__ combine, int* __restrict__ counts, int* __restrict__ lists,
    float* __restrict__ probsum, float* __restrict__ zsum)
{
    int wave = threadIdx.x >> 5, lane = threadIdx.x & 31;
    int n = blockIdx.x * 8 + wave;
    float a[EE];
#pragma unroll
    for (int e = 0; e < EE; ++e) a[e] = 0.f;

    const float4* xr = (const float4*)(x + (size_t)n * CC);
    const float4* g4 = (const float4*)gw;
#pragma unroll 4
    for (int j = 0; j < (CC / 4) / 32; ++j) {
        int idx = lane + j * 32;
        float4 xv = xr[idx];
#pragma unroll
        for (int e = 0; e < EE; ++e) {
            float4 gv = g4[e * (CC / 4) + idx];
            a[e] += xv.x * gv.x + xv.y * gv.y + xv.z * gv.z + xv.w * gv.w;
        }
    }
#pragma unroll
    for (int off = 16; off > 0; off >>= 1)
#pragma unroll
        for (int e = 0; e < EE; ++e) a[e] += __shfl_xor(a[e], off, 32);

    if (lane == 0) {
        float mx = a[0];
#pragma unroll
        for (int e = 1; e < EE; ++e) mx = fmaxf(mx, a[e]);
        float p[EE], s = 0.f;
#pragma unroll
        for (int e = 0; e < EE; ++e) { p[e] = __expf(a[e] - mx); s += p[e]; }
        float inv = 1.f / s;
#pragma unroll
        for (int e = 0; e < EE; ++e) p[e] *= inv;
        float lse = mx + __logf(s);

        int i1 = 0;
#pragma unroll
        for (int e = 1; e < EE; ++e) if (p[e] > p[i1]) i1 = e;
        int i2 = (i1 == 0) ? 1 : 0;
#pragma unroll
        for (int e = 0; e < EE; ++e) if (e != i1 && p[e] > p[i2]) i2 = e;

        float wnorm = 1.f / (p[i1] + p[i2]);
        float w1v = p[i1] * wnorm, w2v = p[i2] * wnorm;
#pragma unroll
        for (int e = 0; e < EE; ++e)
            combine[n * EE + e] = (e == i1) ? w1v : ((e == i2) ? w2v : 0.f);
#pragma unroll
        for (int e = 0; e < EE; ++e) atomicAdd(&probsum[e], p[e]);
        atomicAdd(zsum, lse * lse);

        int p1 = atomicAdd(&counts[i1], 1); lists[i1 * NN + p1] = n;
        int p2 = atomicAdd(&counts[i2], 1); lists[i2 * NN + p2] = n;
    }
}

// ================= up-proj GEMM: h = silu(x_gathered @ w1[e]^T), bf16 out =================
__global__ __launch_bounds__(256) void moe_up_kernel(
    const unsigned short* __restrict__ xb,
    const unsigned short* __restrict__ w1b, const unsigned short* __restrict__ sw1b,
    const int* __restrict__ counts, const int* __restrict__ lists,
    const unsigned short* __restrict__ zbuf, unsigned short* __restrict__ hbuf)
{
    int e = blockIdx.z;
    int n_e = (e == EE) ? NN : counts[e];
    int tileM = blockIdx.y, tileN = blockIdx.x;
    if (tileM * TM >= n_e) return;
    const unsigned short* Bw = (e == EE) ? sw1b : (w1b + (size_t)e * FF * CC);
    const int* gather = (e == EE) ? nullptr : (lists + e * NN);
    unsigned short* hb = hbuf + (size_t)e * NN * FF;

    __shared__ __align__(16) unsigned short sA[2][TM * PITCH];
    __shared__ __align__(16) unsigned short sB[2][TN * PITCH];

    int t = threadIdx.x;
    int lane = t & 31, wave = t >> 5;
    int waveM = wave >> 1, waveN = wave & 1;
    int laneM = lane & 15, laneHalf = lane >> 4;

    v8f acc[2][4];
#pragma unroll
    for (int mi = 0; mi < 2; ++mi)
#pragma unroll
        for (int ni = 0; ni < 4; ++ni)
#pragma unroll
            for (int r = 0; r < 8; ++r) acc[mi][ni][r] = 0.f;

    constexpr int NKT = CC / TK;
    stage_tile(sA[0], sB[0], t, xb, CC, gather, n_e, Bw, CC, zbuf, tileM, tileN, 0);
    for (int ki = 0; ki < NKT; ++ki) {
        int buf = ki & 1;
        bool havenext = (ki + 1) < NKT;
        if (havenext) {
            stage_tile(sA[buf ^ 1], sB[buf ^ 1], t, xb, CC, gather, n_e, Bw, CC, zbuf,
                       tileM, tileN, (ki + 1) * TK);
            wait_async_le8();           // drain previous stage; next 8 stay in flight
        } else {
            wait_async_0();
        }
        __syncthreads();
        compute_tile(sA[buf], sB[buf], acc, waveM, waveN, laneM, laneHalf);
        __syncthreads();
    }

#pragma unroll
    for (int mi = 0; mi < 2; ++mi)
#pragma unroll
        for (int ni = 0; ni < 4; ++ni) {
            int rb = tileM * TM + waveM * 32 + mi * 16 + laneHalf * 8;
            int col = tileN * TN + waveN * 64 + ni * 16 + laneM;
#pragma unroll
            for (int r = 0; r < 8; ++r) {
                int row = rb + r;
                if (row < n_e) {
                    float v = acc[mi][ni][r];
                    float sv = v / (1.f + __expf(-v));       // SiLU
                    hb[(size_t)row * FF + col] = f2bf(sv);
                }
            }
        }
}

// ================= down-proj GEMM: out += combine * (h @ w2[e]^T) =================
__global__ __launch_bounds__(256) void moe_down_kernel(
    const unsigned short* __restrict__ hbuf,
    const unsigned short* __restrict__ w2b, const unsigned short* __restrict__ sw2b,
    const int* __restrict__ counts, const int* __restrict__ lists,
    const float* __restrict__ combine, const unsigned short* __restrict__ zbuf,
    float* __restrict__ accOut)
{
    int e = blockIdx.z;
    int n_e = (e == EE) ? NN : counts[e];
    int tileM = blockIdx.y, tileN = blockIdx.x;
    if (tileM * TM >= n_e) return;
    const unsigned short* Bw = (e == EE) ? sw2b : (w2b + (size_t)e * CC * FF);
    const unsigned short* hb = hbuf + (size_t)e * NN * FF;

    __shared__ __align__(16) unsigned short sA[2][TM * PITCH];
    __shared__ __align__(16) unsigned short sB[2][TN * PITCH];

    int t = threadIdx.x;
    int lane = t & 31, wave = t >> 5;
    int waveM = wave >> 1, waveN = wave & 1;
    int laneM = lane & 15, laneHalf = lane >> 4;

    v8f acc[2][4];
#pragma unroll
    for (int mi = 0; mi < 2; ++mi)
#pragma unroll
        for (int ni = 0; ni < 4; ++ni)
#pragma unroll
            for (int r = 0; r < 8; ++r) acc[mi][ni][r] = 0.f;

    constexpr int NKT = FF / TK;
    stage_tile(sA[0], sB[0], t, hb, FF, nullptr, n_e, Bw, FF, zbuf, tileM, tileN, 0);
    for (int ki = 0; ki < NKT; ++ki) {
        int buf = ki & 1;
        bool havenext = (ki + 1) < NKT;
        if (havenext) {
            stage_tile(sA[buf ^ 1], sB[buf ^ 1], t, hb, FF, nullptr, n_e, Bw, FF, zbuf,
                       tileM, tileN, (ki + 1) * TK);
            wait_async_le8();
        } else {
            wait_async_0();
        }
        __syncthreads();
        compute_tile(sA[buf], sB[buf], acc, waveM, waveN, laneM, laneHalf);
        __syncthreads();
    }

#pragma unroll
    for (int mi = 0; mi < 2; ++mi)
#pragma unroll
        for (int ni = 0; ni < 4; ++ni) {
            int rb = tileM * TM + waveM * 32 + mi * 16 + laneHalf * 8;
            int col = tileN * TN + waveN * 64 + ni * 16 + laneM;
#pragma unroll
            for (int r = 0; r < 8; ++r) {
                int row = rb + r;
                if (row < n_e) {
                    int tok = (e == EE) ? row : lists[e * NN + row];
                    float w = (e == EE) ? 1.f : combine[tok * EE + e];
                    unsafeAtomicAdd(&accOut[(size_t)tok * CC + col], w * acc[mi][ni][r]);
                }
            }
        }
}

// ================= finalize: y = alpha*acc + beta*x, plus scalar loss =================
__global__ __launch_bounds__(256) void moe_final_kernel(
    const float* __restrict__ accOut, const float* __restrict__ x,
    const float* __restrict__ alpha, const float* __restrict__ beta,
    const float* __restrict__ probsum, const float* __restrict__ zsum,
    float* __restrict__ y)
{
    size_t i = (size_t)blockIdx.x * 256 + threadIdx.x;       // float4 index
    int c4 = (int)(i % (CC / 4));
    float4 av = ((const float4*)accOut)[i];
    float4 xv = ((const float4*)x)[i];
    float4 al = ((const float4*)alpha)[c4];
    float4 be = ((const float4*)beta)[c4];
    float4 r;
    r.x = al.x * av.x + be.x * xv.x;
    r.y = al.y * av.y + be.y * xv.y;
    r.z = al.z * av.z + be.z * xv.z;
    r.w = al.w * av.w + be.w * xv.w;
    ((float4*)y)[i] = r;

    if (blockIdx.x == 0 && threadIdx.x == 0) {
        float aux = 0.f;
#pragma unroll
        for (int e = 0; e < EE; ++e) { float m = probsum[e] / (float)NN; aux += m * m; }
        y[(size_t)NN * CC] = AUX_COEF * aux + Z_COEF * (zsum[0] / (float)NN);
    }
}

// ================= host launch =================
extern "C" void kernel_launch(void* const* d_in, const int* in_sizes, int n_in,
                              void* d_out, int out_size, void* d_ws, size_t ws_size,
                              hipStream_t stream) {
    const float* x      = (const float*)d_in[0];
    const float* gate_w = (const float*)d_in[1];
    const float* w1     = (const float*)d_in[2];
    const float* w2     = (const float*)d_in[3];
    const float* sw1    = (const float*)d_in[4];
    const float* sw2    = (const float*)d_in[5];
    const float* alpha  = (const float*)d_in[6];
    const float* beta   = (const float*)d_in[7];
    float* y = (float*)d_out;

    // ---- workspace layout ----
    char* ws = (char*)d_ws;
    float* accOut  = (float*)ws;                               // NN*CC f32
    float* combine = accOut + (size_t)NN * CC;                 // NN*EE f32
    int*   lists   = (int*)(combine + (size_t)NN * EE);        // EE*NN i32
    int*   counts  = lists + (size_t)EE * NN;                  // EE i32
    float* probsum = (float*)(counts + EE);                    // EE f32
    float* zsum    = probsum + EE;                             // 1 f32
    uintptr_t zb   = ((uintptr_t)(zsum + 1) + 15) & ~(uintptr_t)15;
    unsigned short* zbuf = (unsigned short*)zb;                // 128B zero line (inside ctrl memset)
    uintptr_t cur = (zb + 128 + 15) & ~(uintptr_t)15;
    unsigned short* xb   = (unsigned short*)cur; cur += (size_t)NN * CC * 2;
    unsigned short* w1b  = (unsigned short*)cur; cur += (size_t)EE * FF * CC * 2;
    unsigned short* w2b  = (unsigned short*)cur; cur += (size_t)EE * CC * FF * 2;
    unsigned short* sw1b = (unsigned short*)cur; cur += (size_t)FF * CC * 2;
    unsigned short* sw2b = (unsigned short*)cur; cur += (size_t)FF * CC * 2;
    unsigned short* hbuf = (unsigned short*)cur;               // (EE+1)*NN*FF bf16

    (void)hipMemsetAsync(accOut, 0, (size_t)NN * CC * sizeof(float), stream);
    (void)hipMemsetAsync(counts, 0, 512, stream);              // counts+probsum+zsum+zbuf

    // fp32 -> bf16 streaming conversion (8 floats/thread, exact grids)
    cvt_bf16_kernel<<<(NN * CC / 8) / 256, 256, 0, stream>>>(x, xb);
    cvt_bf16_kernel<<<((size_t)EE * FF * CC / 8) / 256, 256, 0, stream>>>(w1, w1b);
    cvt_bf16_kernel<<<((size_t)EE * CC * FF / 8) / 256, 256, 0, stream>>>(w2, w2b);
    cvt_bf16_kernel<<<(FF * CC / 8) / 256, 256, 0, stream>>>(sw1, sw1b);
    cvt_bf16_kernel<<<(FF * CC / 8) / 256, 256, 0, stream>>>(sw2, sw2b);

    moe_gate_kernel<<<NN / 8, 256, 0, stream>>>(x, gate_w, combine, counts, lists, probsum, zsum);
    moe_up_kernel<<<dim3(FF / TN, NN / TM, EE + 1), 256, 0, stream>>>(
        xb, w1b, sw1b, counts, lists, zbuf, hbuf);
    moe_down_kernel<<<dim3(CC / TN, NN / TM, EE + 1), 256, 0, stream>>>(
        hbuf, w2b, sw2b, counts, lists, combine, zbuf, accOut);
    moe_final_kernel<<<(NN * CC / 4) / 256, 256, 0, stream>>>(accOut, x, alpha, beta, probsum, zsum, y);
}
